// Net_89558658056637
// MI455X (gfx1250) — matrix operand order repaired and checked by reading, not compiled, not run on hardware
//
#include <hip/hip_runtime.h>
#include <hip/hip_bf16.h>
#include <math.h>

// ---------------------------------------------------------------------------
// GAT 3-layer pipeline for MI455X (gfx1250, wave32, WMMA bf16, async-to-LDS)
// ---------------------------------------------------------------------------

#define NNODES 10000
#define NEDGES 160000
#define F_IN   512

typedef __attribute__((ext_vector_type(16))) __bf16 v16bf;
typedef __attribute__((ext_vector_type(8)))  float  v8f;
typedef __attribute__((ext_vector_type(2)))  int    v2i_t;

union BFFrag { v16bf bf; unsigned int u[8]; };

#if __has_builtin(__builtin_amdgcn_global_load_async_to_lds_b64) && \
    __has_builtin(__builtin_amdgcn_s_wait_asynccnt)
#define HAVE_ASYNC_LDS 1
typedef __attribute__((address_space(1))) v2i_t* g2i_t;   // global v2i*
typedef __attribute__((address_space(3))) v2i_t* l2i_t;   // LDS v2i*
#endif

__device__ __forceinline__ unsigned short f2bf(float f) {
    unsigned int u = __float_as_uint(f);
    u += 0x7FFFu + ((u >> 16) & 1u);          // round-to-nearest-even
    return (unsigned short)(u >> 16);
}

// fp32 -> bf16 bulk convert (activations, once per layer)
__global__ void conv_bf16(const float* __restrict__ in, unsigned short* __restrict__ out, int n) {
    int i = blockIdx.x * blockDim.x + threadIdx.x;
    if (i < n) out[i] = f2bf(in[i]);
}

// pack [Wa | Wb] fp32 -> one bf16 matrix [K, N1+N2]
__global__ void pack_weights(const float* __restrict__ Wa, const float* __restrict__ Wb,
                             unsigned short* __restrict__ out, int K, int N1, int N2) {
    int idx = blockIdx.x * blockDim.x + threadIdx.x;
    int Nc = N1 + N2;
    if (idx >= K * Nc) return;
    int r = idx / Nc, c = idx - r * Nc;
    float v = (c < N1) ? Wa[(size_t)r * N1 + c] : Wb[(size_t)r * N2 + (c - N1)];
    out[idx] = f2bf(v);
}

// ---------------------------------------------------------------------------
// WMMA GEMM: C[M,Nc] = A[M,K] * B[K,Nc]; A,B bf16, C fp32.
// block = 256 threads (8 waves), block tile 128x128, wave tile 32x64, K-step 32
// A tile staged via GLOBAL_LOAD_ASYNC_TO_LDS_B64 (ASYNCcnt path) when available
// ---------------------------------------------------------------------------
__launch_bounds__(256)
__global__ void gemm_bf16_wmma(const unsigned short* __restrict__ A,
                               const unsigned short* __restrict__ B,
                               float* __restrict__ C,
                               int M, int K, int Nc) {
    __shared__ unsigned short As[128][36];   // [m][k], row stride 72B (8B aligned)
    __shared__ unsigned short Bs[128][34];   // transposed: [n][k]

    const int t    = threadIdx.x;
    const int lane = t & 31;
    const int w    = t >> 5;
    const int bm   = blockIdx.y * 128;
    const int bn   = blockIdx.x * 128;
    const int wm   = (w & 3) * 32;           // wave row offset in block tile
    const int wn   = (w >> 2) * 64;          // wave col offset in block tile
    const int sel  = lane >> 4;              // lane half
    const int l15  = lane & 15;

    v8f acc[2][4] = {};

    for (int kt = 0; kt < K; kt += 32) {
        __syncthreads();
        // ---- stage A tile (128x32 bf16): async copy, 8 bytes per lane op ----
        for (int j = 0; j < 4; ++j) {
            int p   = t + j * 256;           // 1024 x B64 ops
            int i   = p >> 3;                // tile row
            int seg = p & 7;                 // 8-byte segment in row
            int gr  = bm + i;
            if (gr < M) {                    // stale LDS rows feed unstored accs only
                unsigned short* gp = const_cast<unsigned short*>(A) + (size_t)gr * K + kt + seg * 4;
#ifdef HAVE_ASYNC_LDS
                __builtin_amdgcn_global_load_async_to_lds_b64(
                    (g2i_t)gp, (l2i_t)&As[i][seg * 4], 0, 0);
#else
                *(unsigned long long*)&As[i][seg * 4] = *(const unsigned long long*)gp;
#endif
            }
        }
        // ---- stage B tile transposed (128x32): bf16 global, pack k-pairs ----
        for (int j = 0; j < 8; ++j) {
            int p  = t + j * 256;            // 2048 pairs
            int n  = p >> 4;
            int kk = (p & 15) * 2;
            unsigned int pack = 0u;
            int gc = bn + n;
            if (gc < Nc) {
                unsigned int lo = B[(size_t)(kt + kk)     * Nc + gc];
                unsigned int hi = B[(size_t)(kt + kk + 1) * Nc + gc];
                pack = lo | (hi << 16);
            }
            *(unsigned int*)&Bs[n][kk] = pack;
        }
#ifdef HAVE_ASYNC_LDS
        __builtin_amdgcn_s_wait_asynccnt(0);
#endif
        __syncthreads();

        // ---- fragment loads per ISA 16-bit A 16x32 / B 32x16 layouts ----
        BFFrag a[2], b[4];
        for (int r = 0; r < 2; ++r) {
            int mm = wm + r * 16 + l15;
            #pragma unroll
            for (int v = 0; v < 8; ++v) {
                int kk = (v < 4 ? v * 2 : 16 + (v - 4) * 2) + sel * 8;
                a[r].u[v] = *(const unsigned int*)&As[mm][kk];
            }
        }
        for (int c = 0; c < 4; ++c) {
            int nn = wn + c * 16 + l15;
            #pragma unroll
            for (int v = 0; v < 8; ++v) {
                int kk = (v < 4 ? v * 2 : 16 + (v - 4) * 2) + sel * 8;
                b[c].u[v] = *(const unsigned int*)&Bs[nn][kk];
            }
        }
        #pragma unroll
        for (int r = 0; r < 2; ++r)
            #pragma unroll
            for (int c = 0; c < 4; ++c)
                acc[r][c] = __builtin_amdgcn_wmma_f32_16x16x32_bf16(
                    false, a[r].bf, false, b[c].bf,
                    (short)0, acc[r][c], false, false);
    }

    // ---- store C (f32 C/D layout: row = v + 8*sel, col = lane&15) ----
    for (int r = 0; r < 2; ++r) {
        for (int c = 0; c < 4; ++c) {
            int n = bn + wn + c * 16 + l15;
            if (n >= Nc) continue;
            #pragma unroll
            for (int v = 0; v < 8; ++v) {
                int mrow = bm + wm + r * 16 + v + sel * 8;
                if (mrow < M) C[(size_t)mrow * Nc + n] = acc[r][c][v];
            }
        }
    }
}

// ---------------------------------------------------------------------------
// attention coefficients: als/ald[n,h] = sum_c h[n,h,c] * a_{s,d}[h,c]
// hfeat lives in the concatenated GEMM output with row stride ld
// ---------------------------------------------------------------------------
__launch_bounds__(256)
__global__ void attn_coef(const float* __restrict__ hs,
                          const float* __restrict__ a_s,
                          const float* __restrict__ a_d,
                          float* __restrict__ als, float* __restrict__ ald,
                          int H, int C, int ld) {
    int gw   = blockIdx.x * 8 + (threadIdx.x >> 5);
    int lane = threadIdx.x & 31;
    if (gw >= NNODES * H) return;
    int node = gw / H, head = gw % H;
    const float* hp = hs + (size_t)node * ld + head * C;
    const float* sp = a_s + head * C;
    const float* dp = a_d + head * C;
    float ss = 0.f, sd = 0.f;
    for (int c = lane; c < C; c += 32) {
        float hv = hp[c];
        ss += hv * sp[c];
        sd += hv * dp[c];
    }
    for (int off = 16; off > 0; off >>= 1) {
        ss += __shfl_xor(ss, off, 32);
        sd += __shfl_xor(sd, off, 32);
    }
    if (lane == 0) { als[gw] = ss; ald[gw] = sd; }
}

// Q[d,h] = sum_c We[d, h*C+c] * a_e[h,c]  (tiny: 6*H values)
__global__ void qprep(const float* __restrict__ We, const float* __restrict__ ae,
                      float* __restrict__ Q, int H, int C) {
    int t = threadIdx.x;
    if (t >= 6 * H) return;
    int d = t / H, h = t % H;
    float s = 0.f;
    for (int c = 0; c < C; ++c)
        s += We[(size_t)d * H * C + h * C + c] * ae[h * C + c];
    Q[d * H + h] = s;
}

__global__ void fillk(float* __restrict__ p, float v, int n) {
    int i = blockIdx.x * blockDim.x + threadIdx.x;
    if (i < n) p[i] = v;
}

__device__ __forceinline__ void atomicMaxF(float* addr, float val) {
    // sign-aware ordered-int trick (init must be very negative)
    if (val >= 0.f) atomicMax((int*)addr, __float_as_int(val));
    else            atomicMin((unsigned int*)addr, __float_as_uint(val));
}

// logits = leakyrelu(als[src]+ald[dst]+eattr·Q, 0.2); segment max into mmax
__global__ void edge_logits(const int* __restrict__ src, const int* __restrict__ dst,
                            const float* __restrict__ eattr,
                            const float* __restrict__ als, const float* __restrict__ ald,
                            const float* __restrict__ Q,
                            float* __restrict__ elog, float* __restrict__ mmax, int H) {
    int e = blockIdx.x * blockDim.x + threadIdx.x;
    if (e >= NEDGES) return;
    int s = src[e], d = dst[e];
    float ea[6];
    #pragma unroll
    for (int j = 0; j < 6; ++j) ea[j] = eattr[e * 6 + j];
    for (int h = 0; h < H; ++h) {
        float l = als[s * H + h] + ald[d * H + h];
        #pragma unroll
        for (int j = 0; j < 6; ++j) l += ea[j] * Q[j * H + h];
        l = (l > 0.f) ? l : 0.2f * l;
        elog[(size_t)e * H + h] = l;
        atomicMaxF(&mmax[d * H + h], l);
    }
}

// ex = exp(logit - max); segment sum into denom (elog overwritten with ex)
__global__ void edge_exp(const int* __restrict__ dst, float* __restrict__ elog,
                         const float* __restrict__ mmax, float* __restrict__ denom, int H) {
    int idx = blockIdx.x * blockDim.x + threadIdx.x;
    if (idx >= NEDGES * H) return;
    int e = idx / H, h = idx - e * H;
    int d = dst[e];
    float ex = __expf(elog[idx] - mmax[d * H + h]);
    elog[idx] = ex;
    atomicAdd(&denom[d * H + h], ex);
}

// gout[dst,h,:] += (ex/denom[dst,h]) * h[src,h,:]  — one wave per (edge,head)
__launch_bounds__(256)
__global__ void edge_scatter(const int* __restrict__ src, const int* __restrict__ dst,
                             const float* __restrict__ elog, const float* __restrict__ denom,
                             const float* __restrict__ hs, float* __restrict__ gout,
                             int H, int C, int ld) {
    long gw  = (long)blockIdx.x * 8 + (threadIdx.x >> 5);
    int lane = threadIdx.x & 31;
    if (gw >= (long)NEDGES * H) return;
    int e = (int)(gw / H), h = (int)(gw % H);
    int s = src[e], d = dst[e];
    float alpha = elog[(size_t)e * H + h] / denom[d * H + h];
    const float* hp = hs   + (size_t)s * ld + h * C;
    float*       op = gout + (size_t)d * H * C + h * C;
    for (int c = lane; c < C; c += 32)
        atomicAdd(&op[c], alpha * hp[c]);
}

// h_next = elu(gat_out + b + skip + lb); skip lives at column HC of hs (stride ld)
__global__ void combine_elu(const float* __restrict__ gout, const float* __restrict__ hs,
                            const float* __restrict__ bias, const float* __restrict__ lbias,
                            float* __restrict__ out, int F, int ld, int skip_off, int total) {
    int i = blockIdx.x * blockDim.x + threadIdx.x;
    if (i >= total) return;
    int n = i / F, f = i - n * F;
    float v = gout[i] + bias[f] + hs[(size_t)n * ld + skip_off + f] + lbias[f];
    out[i] = (v > 0.f) ? v : expm1f(v);
}

// out[n,c] = mean_h gat3[n,h,c] + b3[c] + skip[n,c] + bl3[c]
__global__ void final_mean(const float* __restrict__ gout, const float* __restrict__ hs,
                           const float* __restrict__ b3, const float* __restrict__ bl3,
                           float* __restrict__ out) {
    int i = blockIdx.x * blockDim.x + threadIdx.x;
    if (i >= NNODES * 16) return;
    int n = i >> 4, c = i & 15;
    float s = 0.f;
    #pragma unroll
    for (int h = 0; h < 6; ++h) s += gout[n * 96 + h * 16 + c];
    out[i] = s * (1.f / 6.f) + b3[c] + hs[(size_t)n * 112 + 96 + c] + bl3[c];
}

// ---------------------------------------------------------------------------
// host orchestration
// ---------------------------------------------------------------------------
extern "C" void kernel_launch(void* const* d_in, const int* in_sizes, int n_in,
                              void* d_out, int out_size, void* d_ws, size_t ws_size,
                              hipStream_t stream) {
    (void)in_sizes; (void)n_in; (void)out_size; (void)ws_size;

    const float* x     = (const float*)d_in[0];
    const int*   ei    = (const int*)d_in[1];
    const int*   src   = ei;
    const int*   dst   = ei + NEDGES;
    const float* eattr = (const float*)d_in[2];

    const float *W1 = (const float*)d_in[3],  *as1 = (const float*)d_in[4],
                *ad1 = (const float*)d_in[5], *We1 = (const float*)d_in[6],
                *ae1 = (const float*)d_in[7], *b1  = (const float*)d_in[8],
                *Wl1 = (const float*)d_in[9], *bl1 = (const float*)d_in[10];
    const float *W2 = (const float*)d_in[11], *as2 = (const float*)d_in[12],
                *ad2 = (const float*)d_in[13], *We2 = (const float*)d_in[14],
                *ae2 = (const float*)d_in[15], *b2  = (const float*)d_in[16],
                *Wl2 = (const float*)d_in[17], *bl2 = (const float*)d_in[18];
    const float *W3 = (const float*)d_in[19], *as3 = (const float*)d_in[20],
                *ad3 = (const float*)d_in[21], *We3 = (const float*)d_in[22],
                *ae3 = (const float*)d_in[23], *b3  = (const float*)d_in[24],
                *Wl3 = (const float*)d_in[25], *bl3 = (const float*)d_in[26];

    // workspace layout (float units)
    float* ws = (float*)d_ws;
    float* hs    = ws;                                   // N*2048 concat [gat|skip] f32
    float* gout  = hs   + (size_t)NNODES * 2048;         // N*1024 aggregation acc
    float* hcur  = gout + (size_t)NNODES * 1024;         // N*1024 activations
    float* als   = hcur + (size_t)NNODES * 1024;         // N*8
    float* ald   = als  + (size_t)NNODES * 8;
    float* mmax  = ald  + (size_t)NNODES * 8;
    float* denom = mmax + (size_t)NNODES * 8;
    float* Q     = denom + (size_t)NNODES * 8;           // 64
    float* elog  = Q + 64;                               // E*8
    unsigned short* abf = (unsigned short*)(elog + (size_t)NEDGES * 8);  // N*1024 bf16
    unsigned short* wbf = abf + (size_t)NNODES * 1024;   // 1024*2048 bf16

    #define CDIV(a, b) (((a) + (b) - 1) / (b))

    for (int layer = 0; layer < 3; ++layer) {
        const float *hin, *W, *Wl, *a_s, *a_d, *We, *a_e;
        int K, H, C;
        if (layer == 0)      { hin = x;    K = F_IN; H = 4; C = 256;
                               W = W1; Wl = Wl1; a_s = as1; a_d = ad1; We = We1; a_e = ae1; }
        else if (layer == 1) { hin = hcur; K = 1024; H = 4; C = 256;
                               W = W2; Wl = Wl2; a_s = as2; a_d = ad2; We = We2; a_e = ae2; }
        else                 { hin = hcur; K = 1024; H = 6; C = 16;
                               W = W3; Wl = Wl3; a_s = as3; a_d = ad3; We = We3; a_e = ae3; }
        const int HC   = H * C;                  // gat width (1024 / 1024 / 96)
        const int F2   = (layer == 2) ? 16 : 1024;  // skip width
        const int Ncat = HC + F2;                // 2048 / 2048 / 112

        // fp32 -> bf16 activations (once), pack [W|Wl] -> bf16
        conv_bf16<<<CDIV(NNODES * K, 256), 256, 0, stream>>>(hin, abf, NNODES * K);
        pack_weights<<<CDIV(K * Ncat, 256), 256, 0, stream>>>(W, Wl, wbf, K, HC, F2);

        // fused GEMM: hs[N, Ncat] = A @ [W|Wl]
        dim3 gg(CDIV(Ncat, 128), CDIV(NNODES, 128));
        gemm_bf16_wmma<<<gg, 256, 0, stream>>>(abf, wbf, hs, NNODES, K, Ncat);

        // attention scalars + edge-feature projection
        attn_coef<<<CDIV(NNODES * H, 8), 256, 0, stream>>>(hs, a_s, a_d, als, ald, H, C, Ncat);
        qprep<<<1, 64, 0, stream>>>(We, a_e, Q, H, C);

        // init reductions + output accumulator
        fillk<<<CDIV(NNODES * H, 256), 256, 0, stream>>>(mmax, -3.4028234663852886e38f, NNODES * H);
        fillk<<<CDIV(NNODES * H, 256), 256, 0, stream>>>(denom, 0.f, NNODES * H);
        fillk<<<CDIV(NNODES * HC, 256), 256, 0, stream>>>(gout, 0.f, NNODES * HC);

        // segment softmax over dst neighborhoods
        edge_logits<<<CDIV(NEDGES, 256), 256, 0, stream>>>(src, dst, eattr, als, ald, Q, elog, mmax, H);
        edge_exp<<<CDIV(NEDGES * H, 256), 256, 0, stream>>>(dst, elog, mmax, denom, H);

        // weighted message scatter
        edge_scatter<<<(int)CDIV((long)NEDGES * H, 8), 256, 0, stream>>>(
            src, dst, elog, denom, hs, gout, H, C, Ncat);

        // combine + activation
        if (layer == 0)
            combine_elu<<<CDIV(NNODES * 1024, 256), 256, 0, stream>>>(
                gout, hs, b1, bl1, hcur, 1024, Ncat, HC, NNODES * 1024);
        else if (layer == 1)
            combine_elu<<<CDIV(NNODES * 1024, 256), 256, 0, stream>>>(
                gout, hs, b2, bl2, hcur, 1024, Ncat, HC, NNODES * 1024);
        else
            final_mean<<<CDIV(NNODES * 16, 256), 256, 0, stream>>>(
                gout, hs, b3, bl3, (float*)d_out);
    }
    #undef CDIV
}
